// MultiHeadAttention_29643864277328
// MI455X (gfx1250) — matrix operand order repaired and checked by reading, not compiled, not run on hardware
//
#include <hip/hip_runtime.h>
#include <cstdint>
#include <cstddef>

#define N_TOK   4096
#define EMB_D   1024
#define N_HEADS 16
#define HEAD_D  64

// IGLP scheduling groups: 0x020 = VMEM read, 0x008 = MFMA/WMMA
#define SGB(mask, size) __builtin_amdgcn_sched_group_barrier((mask), (size), 0)

// ---- POD vector types (avoid HIP uint4 ctors inside unions) ----
typedef __attribute__((ext_vector_type(16))) __bf16        v16bf;
typedef __attribute__((ext_vector_type(8)))  float         v8f;
typedef __attribute__((ext_vector_type(4)))  unsigned int  u32x4;

union Frag16 {
    v16bf          bf;
    u32x4          q[2];
    unsigned short us[16];
};

static __device__ __forceinline__ unsigned short f32_bf16(float f) {
    unsigned int u = __float_as_uint(f);
    u += 0x7FFFu + ((u >> 16) & 1u);   // round-to-nearest-even
    return (unsigned short)(u >> 16);
}

static __device__ __forceinline__ v8f wmma_bf16(v16bf a, v16bf b, v8f c) {
    // (neg_a, A, neg_b, B, c_mod, C, reuse_a, reuse_b)
    return __builtin_amdgcn_wmma_f32_16x16x32_bf16(false, a, false, b, (short)0, c,
                                                   false, false);
}

// Wave-local LDS ordering: same wave's lanes exchange P tile through LDS.
static __device__ __forceinline__ void lds_fence() {
    __builtin_amdgcn_wave_barrier();
    asm volatile("s_wait_dscnt 0" ::: "memory");
    __builtin_amdgcn_wave_barrier();
}

// A fragment: 16x32 bf16 tile, rows row0..row0+15, K window k0..k0+31,
// from a row-major (ld elements/row) bf16 matrix.
// ISA layout: lane&15 = row M; lanes<16 hold K {0..7,16..23}, lanes>=16 {8..15,24..31}.
static __device__ __forceinline__ v16bf load_a(const unsigned short* m, size_t ld,
                                               int row0, int k0, int l16, int half) {
    Frag16 f;
    const unsigned short* p = m + (size_t)(row0 + l16) * ld + (size_t)k0 + (size_t)half * 8;
    f.q[0] = *reinterpret_cast<const u32x4*>(p);        // K = k0+half*8 .. +7
    f.q[1] = *reinterpret_cast<const u32x4*>(p + 16);   // K = k0+16+half*8 .. +7
    return f.bf;
}

// B fragment: 32x16 bf16 tile where source is stored "transposed" (one GEMM
// column = one contiguous row of K values): bt[col][k], ld elements/row.
// ISA layout: lane&15 = col N; lanes<16 hold K {0..7,16..23}, lanes>=16 {8..15,24..31}.
static __device__ __forceinline__ v16bf load_bt(const unsigned short* bt, size_t ld,
                                                int col0, int k0, int l16, int half) {
    Frag16 f;
    const unsigned short* p = bt + (size_t)(col0 + l16) * ld + (size_t)k0 + (size_t)half * 8;
    f.q[0] = *reinterpret_cast<const u32x4*>(p);        // K = k0+half*8 .. +7
    f.q[1] = *reinterpret_cast<const u32x4*>(p + 16);   // K = k0+16+half*8 .. +7
    return f.bf;
}

// ------------------------------ conversion kernels ------------------------------

__global__ void k_cvt_bf16(const float* __restrict__ src, unsigned short* __restrict__ dst, int n) {
    int i = blockIdx.x * blockDim.x + threadIdx.x;
    if (i < n) dst[i] = f32_bf16(src[i]);
}

// (H,E,D) fp32 -> (H,D,E) bf16
__global__ void k_cvt_w_hed(const float* __restrict__ w, unsigned short* __restrict__ wt, int n) {
    int i = blockIdx.x * blockDim.x + threadIdx.x;
    if (i >= n) return;
    int d = i % HEAD_D;
    int e = (i / HEAD_D) % EMB_D;
    int h = i / (HEAD_D * EMB_D);
    wt[((size_t)h * HEAD_D + d) * EMB_D + e] = f32_bf16(w[i]);
}

// (K=1024, E=1024) fp32 -> (E, K) bf16
__global__ void k_cvt_wp(const float* __restrict__ w, unsigned short* __restrict__ wt, int n) {
    int i = blockIdx.x * blockDim.x + threadIdx.x;
    if (i >= n) return;
    int e = i % EMB_D;
    int k = i / EMB_D;
    wt[(size_t)e * EMB_D + k] = f32_bf16(w[i]);
}

// ------------------------------ projection GEMM ------------------------------
// out[h] = X(N,E) @ W[h](E,D); X bf16 row-major, W supplied transposed (H,D,E).
// transpose_out==0 -> (H,N,D); ==1 -> (H,D,N)  (used for V so P@V B-frags are contiguous)
// Fully unrolled K loop; sched_group_barrier pins a one-block-ahead pipeline:
//   V10(blk0), [V10(blk k+1), M4(blk k)] ...
__global__ __launch_bounds__(256) void k_proj(const unsigned short* __restrict__ xbf,
                                              const unsigned short* __restrict__ wt,
                                              unsigned short* __restrict__ out,
                                              int transpose_out) {
    int wid  = blockIdx.x * (blockDim.x >> 5) + (threadIdx.x >> 5);
    int lane = threadIdx.x & 31;
    int l16 = lane & 15, half = lane >> 4;
    int h  = wid >> 8;        // 256 row tiles per head
    int qt = wid & 255;
    int row0 = qt * 16;
    const unsigned short* wh = wt + (size_t)h * HEAD_D * EMB_D;

    v8f zero = {0.f, 0.f, 0.f, 0.f, 0.f, 0.f, 0.f, 0.f};
    v8f acc[4] = {zero, zero, zero, zero};

    v16bf aC = load_a(xbf, EMB_D, row0, 0, l16, half);
    v16bf bC[4];
#pragma unroll
    for (int t = 0; t < 4; ++t) bC[t] = load_bt(wh, EMB_D, t * 16, 0, l16, half);
    SGB(0x020, 10);                          // group: block-0 loads (10 x b128)

#pragma unroll
    for (int k0 = 0; k0 < EMB_D; k0 += 32) {
        v16bf aN = aC;
        v16bf bN[4] = {bC[0], bC[1], bC[2], bC[3]};
        if (k0 + 32 < EMB_D) {               // constant per unrolled copy
            aN = load_a(xbf, EMB_D, row0, k0 + 32, l16, half);
#pragma unroll
            for (int t = 0; t < 4; ++t) bN[t] = load_bt(wh, EMB_D, t * 16, k0 + 32, l16, half);
        }
        SGB(0x020, 10);                      // group: block k+1 loads
#pragma unroll
        for (int t = 0; t < 4; ++t) acc[t] = wmma_bf16(aC, bC[t], acc[t]);
        SGB(0x008, 4);                       // group: block k WMMAs
        aC = aN;                             // SSA renaming only (fully unrolled)
#pragma unroll
        for (int t = 0; t < 4; ++t) bC[t] = bN[t];
    }

#pragma unroll
    for (int t = 0; t < 4; ++t) {
#pragma unroll
        for (int r = 0; r < 8; ++r) {
            int row = row0 + r + 8 * half;
            int col = t * 16 + l16;
            unsigned short v = f32_bf16(acc[t][r]);
            if (!transpose_out)
                out[((size_t)h * N_TOK + row) * HEAD_D + col] = v;
            else
                out[((size_t)h * HEAD_D + col) * N_TOK + row] = v;
        }
    }
}

// ------------------------------ flash attention ------------------------------
// One wave per (head, 16-query tile). Q,K: (H,N,D) bf16; Vt: (H,D,N) bf16.
// heads out: (N, H*D) bf16 (concat layout).
// Pipelined: V fragments issued before softmax; next K block preloaded in place
// after S-WMMAs; global_prefetch warms K (2 blocks ahead) and V (1 block ahead).
__global__ __launch_bounds__(256) void k_flash_attn(const unsigned short* __restrict__ Q,
                                                    const unsigned short* __restrict__ K,
                                                    const unsigned short* __restrict__ Vt,
                                                    unsigned short* __restrict__ heads) {
    __shared__ __align__(16) unsigned short ptile_all[8][16 * 32];
    int wslot = threadIdx.x >> 5;
    unsigned short* ptile = ptile_all[wslot];
    int wid  = blockIdx.x * (blockDim.x >> 5) + wslot;
    int lane = threadIdx.x & 31;
    int l16 = lane & 15, half = lane >> 4;
    int h  = wid >> 8;
    int qt = wid & 255;
    int qbase = qt * 16;

    const unsigned short* Qh = Q  + (size_t)h * N_TOK * HEAD_D;
    const unsigned short* Kh = K  + (size_t)h * N_TOK * HEAD_D;
    const unsigned short* Vh = Vt + (size_t)h * HEAD_D * N_TOK;

    // Q A-fragments resident for whole key loop (D=64 -> two K=32 windows)
    v16bf aq0 = load_a(Qh, HEAD_D, qbase, 0,  l16, half);
    v16bf aq1 = load_a(Qh, HEAD_D, qbase, 32, l16, half);

    v8f zero = {0.f, 0.f, 0.f, 0.f, 0.f, 0.f, 0.f, 0.f};
    v8f acc[4] = {zero, zero, zero, zero};     // O tile: 16 x 64
    float mr[8], lr[8];                        // per-row running max / sum
#pragma unroll
    for (int r = 0; r < 8; ++r) { mr[r] = -1e30f; lr[r] = 0.f; }

    const float scale = 0.125f;                // 1/sqrt(64)
    int nkb = qbase / 32 + 1;                  // causal: key blocks covering <= qbase+15

    // prologue: K fragments for block 0   [score tile][d-window]
    v16bf bk00 = load_bt(Kh, HEAD_D, 0,  0,  l16, half);
    v16bf bk01 = load_bt(Kh, HEAD_D, 0,  32, l16, half);
    v16bf bk10 = load_bt(Kh, HEAD_D, 16, 0,  l16, half);
    v16bf bk11 = load_bt(Kh, HEAD_D, 16, 32, l16, half);

    for (int kb = 0; kb < nkb; ++kb) {
        int kbase = kb * 32;

        // V fragments for this block: issue early, consumed after softmax
        v16bf bv[4];
#pragma unroll
        for (int t = 0; t < 4; ++t) bv[t] = load_bt(Vh, N_TOK, t * 16, kbase, l16, half);

        // S = Q * K^T  (two 16x16 score tiles, accumulated over d=0..63)
        v8f s[2];
        s[0] = wmma_bf16(aq1, bk01, wmma_bf16(aq0, bk00, zero));
        s[1] = wmma_bf16(aq1, bk11, wmma_bf16(aq0, bk10, zero));

        // preload next K block in place; latency hides under the softmax VALU work
        if (kb + 1 < nkb) {
            int kn = kbase + 32;
            bk00 = load_bt(Kh, HEAD_D, kn,      0,  l16, half);
            bk01 = load_bt(Kh, HEAD_D, kn,      32, l16, half);
            bk10 = load_bt(Kh, HEAD_D, kn + 16, 0,  l16, half);
            bk11 = load_bt(Kh, HEAD_D, kn + 16, 32, l16, half);
            // prefetch K two blocks ahead (one 128B row per lane covers 32 rows)
            if (kb + 2 < nkb)
                __builtin_prefetch(Kh + (size_t)(kbase + 64 + lane) * HEAD_D, 0, 3);
            // prefetch V one block ahead
#pragma unroll
            for (int t = 0; t < 4; ++t)
                __builtin_prefetch(Vh + (size_t)(t * 16 + l16) * N_TOK + kn, 0, 3);
        }

        // scale + causal mask + online softmax (row = r + 8*half, col = t*16 + l16)
#pragma unroll
        for (int r = 0; r < 8; ++r) {
            int q = qbase + r + 8 * half;
            float s0 = s[0][r] * scale;
            float s1 = s[1][r] * scale;
            if (kbase + l16 > q)      s0 = -1e30f;
            if (kbase + 16 + l16 > q) s1 = -1e30f;
            float mn = fmaxf(mr[r], fmaxf(s0, s1));
#pragma unroll
            for (int d = 1; d < 16; d <<= 1) mn = fmaxf(mn, __shfl_xor(mn, d, 32));
            float alpha = __expf(mr[r] - mn);
            mr[r] = mn;
            float p0 = __expf(s0 - mn);
            float p1 = __expf(s1 - mn);
            float ps = p0 + p1;
#pragma unroll
            for (int d = 1; d < 16; d <<= 1) ps += __shfl_xor(ps, d, 32);
            lr[r] = lr[r] * alpha + ps;
#pragma unroll
            for (int t = 0; t < 4; ++t) acc[t][r] *= alpha;
            // stage P (bf16) through LDS to re-layout C-frag -> A-frag
            ptile[(r + 8 * half) * 32 + l16]      = f32_bf16(p0);
            ptile[(r + 8 * half) * 32 + 16 + l16] = f32_bf16(p1);
        }
        lds_fence();
        Frag16 pa;
        const unsigned short* pp = ptile + l16 * 32 + half * 8;
        pa.q[0] = *reinterpret_cast<const u32x4*>(pp);
        pa.q[1] = *reinterpret_cast<const u32x4*>(pp + 16);
        // O += P * V   (V fragments already in flight since loop top)
#pragma unroll
        for (int t = 0; t < 4; ++t) acc[t] = wmma_bf16(pa.bf, bv[t], acc[t]);
        lds_fence();   // keep next iteration's P stores ordered after these loads
    }
    // epilogue: normalize and write concat-head layout (N, H*D)
#pragma unroll
    for (int r = 0; r < 8; ++r) {
        float inv = 1.0f / lr[r];
        int row = qbase + r + 8 * half;
#pragma unroll
        for (int t = 0; t < 4; ++t)
            heads[(size_t)row * (N_HEADS * HEAD_D) + h * HEAD_D + t * 16 + l16] =
                f32_bf16(acc[t][r] * inv);
    }
}

// ------------------------------ output pooling GEMM ------------------------------
// out(N,E) fp32 = heads(N,1024) bf16 @ W_pool(1024,E); wpt stored (E,1024).
// Fully unrolled + sched_group_barrier pipeline like k_proj.
__global__ __launch_bounds__(256) void k_pool(const unsigned short* __restrict__ heads,
                                              const unsigned short* __restrict__ wpt,
                                              float* __restrict__ out) {
    int wid  = blockIdx.x * (blockDim.x >> 5) + (threadIdx.x >> 5);
    int lane = threadIdx.x & 31;
    int l16 = lane & 15, half = lane >> 4;
    int ct = wid & 15;          // 16 column tiles of 64
    int rt = wid >> 4;          // 256 row tiles of 16
    int row0 = rt * 16, col0 = ct * 64;
    const int KD = N_HEADS * HEAD_D;   // 1024

    v8f zero = {0.f, 0.f, 0.f, 0.f, 0.f, 0.f, 0.f, 0.f};
    v8f acc[4] = {zero, zero, zero, zero};

    v16bf aC = load_a(heads, KD, row0, 0, l16, half);
    v16bf bC[4];
#pragma unroll
    for (int t = 0; t < 4; ++t) bC[t] = load_bt(wpt, KD, col0 + t * 16, 0, l16, half);
    SGB(0x020, 10);

#pragma unroll
    for (int k0 = 0; k0 < KD; k0 += 32) {
        v16bf aN = aC;
        v16bf bN[4] = {bC[0], bC[1], bC[2], bC[3]};
        if (k0 + 32 < KD) {
            aN = load_a(heads, KD, row0, k0 + 32, l16, half);
#pragma unroll
            for (int t = 0; t < 4; ++t) bN[t] = load_bt(wpt, KD, col0 + t * 16, k0 + 32, l16, half);
        }
        SGB(0x020, 10);
#pragma unroll
        for (int t = 0; t < 4; ++t) acc[t] = wmma_bf16(aC, bC[t], acc[t]);
        SGB(0x008, 4);
        aC = aN;
#pragma unroll
        for (int t = 0; t < 4; ++t) bC[t] = bN[t];
    }

#pragma unroll
    for (int r = 0; r < 8; ++r) {
        int row = row0 + r + 8 * half;
#pragma unroll
        for (int t = 0; t < 4; ++t)
            out[(size_t)row * EMB_D + col0 + t * 16 + l16] = acc[t][r];
    }
}

// ------------------------------ launch ------------------------------

extern "C" void kernel_launch(void* const* d_in, const int* in_sizes, int n_in,
                              void* d_out, int out_size, void* d_ws, size_t ws_size,
                              hipStream_t stream) {
    (void)in_sizes; (void)n_in; (void)out_size; (void)ws_size;
    const float* x  = (const float*)d_in[0];   // (N, E)
    const float* wq = (const float*)d_in[1];   // (H, E, D)
    const float* wk = (const float*)d_in[2];
    const float* wv = (const float*)d_in[3];
    const float* wp = (const float*)d_in[4];   // (H*D, E)
    float* out = (float*)d_out;                // (N, E)

    unsigned short* ws = (unsigned short*)d_ws;
    size_t off = 0;
    unsigned short* xbf  = ws + off; off += (size_t)N_TOK * EMB_D;            // 4M elems
    unsigned short* wqt  = ws + off; off += (size_t)N_HEADS * HEAD_D * EMB_D; // 1M
    unsigned short* wkt  = ws + off; off += (size_t)N_HEADS * HEAD_D * EMB_D;
    unsigned short* wvt  = ws + off; off += (size_t)N_HEADS * HEAD_D * EMB_D;
    unsigned short* wpt  = ws + off; off += (size_t)EMB_D * EMB_D;            // 1M
    unsigned short* Qb   = ws + off; off += (size_t)N_HEADS * N_TOK * HEAD_D; // 4M
    unsigned short* Kb   = ws + off; off += (size_t)N_HEADS * N_TOK * HEAD_D;
    unsigned short* Vtb  = ws + off; off += (size_t)N_HEADS * N_TOK * HEAD_D;
    unsigned short* hdsb = ws + off; off += (size_t)N_TOK * N_HEADS * HEAD_D;

    const int T = 256;
    int n_x = N_TOK * EMB_D;
    int n_w = N_HEADS * EMB_D * HEAD_D;
    int n_p = EMB_D * EMB_D;

    k_cvt_bf16 <<<(n_x + T - 1) / T, T, 0, stream>>>(x,  xbf, n_x);
    k_cvt_w_hed<<<(n_w + T - 1) / T, T, 0, stream>>>(wq, wqt, n_w);
    k_cvt_w_hed<<<(n_w + T - 1) / T, T, 0, stream>>>(wk, wkt, n_w);
    k_cvt_w_hed<<<(n_w + T - 1) / T, T, 0, stream>>>(wv, wvt, n_w);
    k_cvt_wp   <<<(n_p + T - 1) / T, T, 0, stream>>>(wp, wpt, n_p);

    // 4096 waves each (16 heads x 256 row tiles), 8 waves/block
    k_proj<<<512, 256, 0, stream>>>(xbf, wqt, Qb,  0);
    k_proj<<<512, 256, 0, stream>>>(xbf, wkt, Kb,  0);
    k_proj<<<512, 256, 0, stream>>>(xbf, wvt, Vtb, 1);   // V stored (H, D, N)

    k_flash_attn<<<512, 256, 0, stream>>>(Qb, Kb, Vtb, hdsb);

    k_pool<<<512, 256, 0, stream>>>(hdsb, wpt, out);
}
